// KnowledgeGraph_73237782331452
// MI455X (gfx1250) — matrix operand order repaired and checked by reading, not compiled, not run on hardware
//
#include <hip/hip_runtime.h>

// ---------------------------------------------------------------------------
// KGAT-style neighbor aggregation, fused for MI455X (gfx1250, wave32, WMMA).
// GEMMs run as v_wmma_f32_16x16x32_bf16 (f32 accumulate); gathers are the
// bandwidth-critical path and are staged through LDS as bf16.
// ---------------------------------------------------------------------------

typedef __attribute__((ext_vector_type(16))) __bf16 v16bf;
typedef __attribute__((ext_vector_type(8)))  __bf16 v8bf;
typedef __attribute__((ext_vector_type(4)))  __bf16 v4bf;
typedef __attribute__((ext_vector_type(8)))  float  v8f;

#define D_    200         // embedding dim
#define K_    10          // neighbors per entity
#define TB    8           // queries per workgroup
#define ROWS  (TB * K_)   // 80 agg rows per workgroup
#define KDIM  400         // 2*D (GEMM reduction dim)
#define KPAD  416         // KDIM padded to multiple of 32
#define NPAD  208         // D padded to multiple of 16
#define NKT   (KPAD / 32) // 13 k-steps per tile
#define NNT   (NPAD / 16) // 13 n-tiles
#define NMT   (ROWS / 16) // 5 m-tiles

// ---- Weight prep: WT[n][k] = bf16(W[k][n]), zero padded to [208][416] ------
__global__ __launch_bounds__(256) void prep_weights(const float* __restrict__ Wagg,
                                                    const float* __restrict__ Wself,
                                                    __bf16* __restrict__ wtA,
                                                    __bf16* __restrict__ wtS) {
  int i = blockIdx.x * blockDim.x + threadIdx.x;
  if (i >= NPAD * KPAD) return;
  int n = i / KPAD;
  int k = i % KPAD;
  bool ok = (n < D_) && (k < KDIM);
  wtA[i] = ok ? (__bf16)Wagg[k * D_ + n]  : (__bf16)0.0f;
  wtS[i] = ok ? (__bf16)Wself[k * D_ + n] : (__bf16)0.0f;
}

// ---- WMMA fragment loaders (CDNA5 16-bit operand layouts, wave32) ----------
// A (16x32 bf16): lane<16 holds row=lane, K-chunks {kb+0..7, kb+16..23};
//                 lane>=16 holds row=lane-16, K-chunks {kb+8..15, kb+24..31}.
__device__ __forceinline__ v16bf load_a_frag(const __bf16* base, int row,
                                             int kbase, int lane) {
  const __bf16* p = base + row * KPAD + kbase + ((lane & 16) ? 8 : 0);
  v8bf lo = *(const v8bf*)p;          // 16B chunk
  v8bf hi = *(const v8bf*)(p + 16);   // 16B chunk, +32B
  return __builtin_shufflevector(lo, hi, 0, 1, 2, 3, 4, 5, 6, 7,
                                         8, 9, 10, 11, 12, 13, 14, 15);
}

// B (32x16 bf16): lane holds column n = lane&15; lane<16 -> K-chunks {kb+0..7, kb+16..23},
// lane>=16 -> K-chunks {kb+8..15, kb+24..31} (two 16B chunks of W^T row n).
__device__ __forceinline__ v16bf load_b_frag(const __bf16* __restrict__ wt,
                                             int n, int kbase, int lane) {
  const __bf16* p = wt + n * KPAD + kbase + ((lane & 16) ? 8 : 0);
  v8bf lo = *(const v8bf*)p;
  v8bf hi = *(const v8bf*)(p + 16);
  return __builtin_shufflevector(lo, hi, 0, 1, 2, 3, 4, 5, 6, 7,
                                         8, 9, 10, 11, 12, 13, 14, 15);
}

// ---- Fused kernel: gather -> GEMM1 -> attention/pool -> GEMM2 --------------
__global__ __launch_bounds__(256) void kgat_fused(
    const int*   __restrict__ entity_ids,
    const int*   __restrict__ neigh_rels,
    const int*   __restrict__ neigh_ents,
    const float* __restrict__ entity_emb,
    const float* __restrict__ relation_emb,
    const float* __restrict__ b_agg,
    const float* __restrict__ w_u,
    const float* __restrict__ b_u,
    const float* __restrict__ b_self,
    const __bf16* __restrict__ WT_agg,
    const __bf16* __restrict__ WT_self,
    float* __restrict__ out) {
  __shared__ __bf16 sA[ROWS * KPAD];     // concat rows, bf16         (65.0 KB)
  __shared__ float  sAGG[ROWS * NPAD];   // agg rows, f32             (65.0 KB)
  __shared__ __bf16 sSelf[16 * KPAD];    // [self | pooled], bf16     (13.0 KB)
  __shared__ float  sLogit[ROWS];

  const int tid  = threadIdx.x;
  const int lane = tid & 31;
  const int wave = tid >> 5;             // 8 waves of 32 (wave32)
  const int b0   = blockIdx.x * TB;

  // ---- Phase 1: gather neighbor rows, convert f32->bf16 into LDS ----------
  for (int r = wave; r < ROWS; r += 8) {
    const int q   = r / K_;
    const int kk  = r % K_;
    const int eid = entity_ids[b0 + q];
    const int rid = neigh_rels[eid * K_ + kk];
    const int nid = neigh_ents[eid * K_ + kk];
    const float4* relp = (const float4*)(relation_emb + (long)rid * D_);
    const float4* entp = (const float4*)(entity_emb  + (long)nid * D_);
    for (int c4 = lane; c4 < KPAD / 4; c4 += 32) {
      float4 v;
      if (c4 < 50)       v = relp[c4];
      else if (c4 < 100) v = entp[c4 - 50];
      else               v = make_float4(0.f, 0.f, 0.f, 0.f);  // K padding
      v4bf h;
      h[0] = (__bf16)v.x; h[1] = (__bf16)v.y;
      h[2] = (__bf16)v.z; h[3] = (__bf16)v.w;
      *(v4bf*)(sA + r * KPAD + c4 * 4) = h;
    }
  }
  // zero the second-GEMM A matrix (pad rows 8..15 and K padding)
  for (int i = tid; i < 16 * KPAD / 4; i += 256) {
    v4bf z = {(__bf16)0.f, (__bf16)0.f, (__bf16)0.f, (__bf16)0.f};
    *(v4bf*)(sSelf + i * 4) = z;
  }
  __syncthreads();

  // self embeddings into sSelf rows 0..TB-1, cols 0..199 (read in phase 4)
  {
    const int eid = entity_ids[b0 + wave];  // wave < TB == 8
    const float4* sp = (const float4*)(entity_emb + (long)eid * D_);
    for (int c4 = lane; c4 < 50; c4 += 32) {
      float4 v = sp[c4];
      v4bf h;
      h[0] = (__bf16)v.x; h[1] = (__bf16)v.y;
      h[2] = (__bf16)v.z; h[3] = (__bf16)v.w;
      *(v4bf*)(sSelf + wave * KPAD + c4 * 4) = h;
    }
  }

  // ---- Phase 2: AGG[80x200] = concat[80x400] @ W_agg[400x200] (bf16 WMMA) -
  for (int t = wave; t < NMT * NNT; t += 8) {
    const int mt   = t / NNT;
    const int nt   = t % NNT;
    const int arow = mt * 16 + (lane & 15);
    const int n    = nt * 16 + (lane & 15);
    v8f acc = {};
    for (int ks = 0; ks < NKT; ++ks) {
      v16bf a = load_a_frag(sA, arow, ks * 32, lane);
      v16bf b = load_b_frag(WT_agg, n, ks * 32, lane);
      acc = __builtin_amdgcn_wmma_f32_16x16x32_bf16(false, a, false, b,
                                                    (short)0, acc, false, false);
    }
    const float bias  = (n < D_) ? b_agg[n] : 0.f;
    const int   mbase = mt * 16 + ((lane & 16) ? 8 : 0);  // C layout: V_j -> row
#pragma unroll
    for (int j = 0; j < 8; ++j)
      sAGG[(mbase + j) * NPAD + n] = acc[j] + bias;
  }
  __syncthreads();

  // ---- Phase 3a: logits[r] = tanh(agg_r) . w_u + b_u ----------------------
  for (int j = 0; j < K_; ++j) {
    const int r = wave * K_ + j;
    float s = 0.f;
    for (int c = lane; c < D_; c += 32)
      s += tanhf(sAGG[r * NPAD + c]) * w_u[c];
#pragma unroll
    for (int off = 16; off; off >>= 1) s += __shfl_xor(s, off, 32);
    if (lane == 0) sLogit[r] = s + b_u[0];
  }
  __syncthreads();

  // ---- Phase 3b: softmax over K, pooled -> sSelf cols 200..399 ------------
  {
    const int q = wave;  // one query per wave
    float m = -1e30f;
#pragma unroll
    for (int j = 0; j < K_; ++j) m = fmaxf(m, sLogit[q * K_ + j]);
    float e[K_];
    float Z = 0.f;
#pragma unroll
    for (int j = 0; j < K_; ++j) {
      e[j] = __expf(sLogit[q * K_ + j] - m);
      Z += e[j];
    }
    const float invZ = 1.f / Z;
    for (int c = lane; c < D_; c += 32) {
      float p = 0.f;
#pragma unroll
      for (int j = 0; j < K_; ++j) p += e[j] * sAGG[(q * K_ + j) * NPAD + c];
      sSelf[q * KPAD + D_ + c] = (__bf16)(p * invZ);
    }
  }
  __syncthreads();

  // ---- Phase 4: out = tanh([self|pooled][8x400] @ W_self + b_self) --------
  for (int nt = wave; nt < NNT; nt += 8) {
    const int n = nt * 16 + (lane & 15);
    v8f acc = {};
    for (int ks = 0; ks < NKT; ++ks) {
      v16bf a = load_a_frag(sSelf, (lane & 15), ks * 32, lane);
      v16bf b = load_b_frag(WT_self, n, ks * 32, lane);
      acc = __builtin_amdgcn_wmma_f32_16x16x32_bf16(false, a, false, b,
                                                    (short)0, acc, false, false);
    }
    if (lane < 16 && n < D_) {           // lanes 0..15 hold rows M=0..7 (real queries)
      const float bias = b_self[n];
#pragma unroll
      for (int j = 0; j < 8; ++j)
        out[(long)(b0 + j) * D_ + n] = tanhf(acc[j] + bias);
    }
  }
}

// ---------------------------------------------------------------------------
extern "C" void kernel_launch(void* const* d_in, const int* in_sizes, int n_in,
                              void* d_out, int out_size, void* d_ws, size_t ws_size,
                              hipStream_t stream) {
  const int*   entity_ids   = (const int*)d_in[0];
  const int*   neigh_rels   = (const int*)d_in[1];
  const int*   neigh_ents   = (const int*)d_in[2];
  const float* entity_emb   = (const float*)d_in[3];
  const float* relation_emb = (const float*)d_in[4];
  const float* W_agg        = (const float*)d_in[5];
  const float* b_agg        = (const float*)d_in[6];
  const float* w_u          = (const float*)d_in[7];
  const float* b_u          = (const float*)d_in[8];
  const float* W_self       = (const float*)d_in[9];
  const float* b_self       = (const float*)d_in[10];
  float* out = (float*)d_out;
  const int B = in_sizes[0];

  __bf16* WT_agg  = (__bf16*)d_ws;
  __bf16* WT_self = (__bf16*)((char*)d_ws + (size_t)NPAD * KPAD * sizeof(__bf16));

  const int total = NPAD * KPAD;
  prep_weights<<<(total + 255) / 256, 256, 0, stream>>>(W_agg, W_self, WT_agg, WT_self);
  kgat_fused<<<B / TB, 256, 0, stream>>>(entity_ids, neigh_rels, neigh_ents,
                                         entity_emb, relation_emb, b_agg, w_u, b_u,
                                         b_self, WT_agg, WT_self, out);
}
